// CNN_12154757447795
// MI455X (gfx1250) — matrix operand order repaired and checked by reading, not compiled, not run on hardware
//
#include <hip/hip_runtime.h>
#include <hip/hip_bf16.h>

typedef __attribute__((ext_vector_type(16))) __bf16   bf16x16;
typedef __attribute__((ext_vector_type(8)))  __bf16   bf16x8;
typedef __attribute__((ext_vector_type(8)))  float    f32x8;
typedef __attribute__((ext_vector_type(8)))  unsigned u32x8;

__device__ __forceinline__ f32x8 wmma_bf16(bf16x16 a, bf16x16 b, f32x8 c) {
  return __builtin_amdgcn_wmma_f32_16x16x32_bf16(false, a, false, b, (short)0, c,
                                                 false, false);
}

// ---------------------------------------------------------------------------
// Pack OIHW fp32 weights -> bf16 [tap(10)][O][I], tap = ky*3+kx, tap 9 = zeros
// (K pad: 9 taps x 16 ch = 144 -> 160 = 5 x K32 WMMA steps; the zero tap lets
// B-operands alias tap 8 for K=16..31 of the last step with no masking).
// ---------------------------------------------------------------------------
__global__ __launch_bounds__(256) void pack_w(const float* __restrict__ src,
                                              __bf16* __restrict__ dst,
                                              int O, int I) {
  int idx = blockIdx.x * blockDim.x + threadIdx.x;
  int total = 10 * O * I;
  if (idx >= total) return;
  int i = idx % I;
  int o = (idx / I) % O;
  int t = idx / (I * O);
  float v = (t < 9) ? src[(o * I + i) * 9 + t] : 0.0f;
  dst[idx] = (__bf16)v;
}

// ---------------------------------------------------------------------------
// Head conv: 1->16 ch, 3x3, pad 1, stride s. fp32 NCHW in -> bf16 NHWC16 out.
// ---------------------------------------------------------------------------
__global__ __launch_bounds__(256) void head_conv(const float* __restrict__ x,
                                                 const float* __restrict__ w,   // [16][9]
                                                 const float* __restrict__ b,   // [16]
                                                 __bf16* __restrict__ out,      // NHWC16
                                                 int N, int Hi, int Wi,
                                                 int stride, int Ho, int Wo) {
  int idx = blockIdx.x * blockDim.x + threadIdx.x;
  int tot = N * Ho * Wo;
  if (idx >= tot) return;
  int xo = idx % Wo;
  int t  = idx / Wo;
  int yo = t % Ho;
  int n  = t / Ho;

  const float* img = x + (size_t)n * Hi * Wi;
  float acc[16];
#pragma unroll
  for (int o = 0; o < 16; ++o) acc[o] = b[o];
  for (int ky = 0; ky < 3; ++ky) {
    int yi = yo * stride + ky - 1;
    if (yi < 0 || yi >= Hi) continue;
    for (int kx = 0; kx < 3; ++kx) {
      int xi = xo * stride + kx - 1;
      if (xi < 0 || xi >= Wi) continue;
      float xv = img[yi * Wi + xi];
#pragma unroll
      for (int o = 0; o < 16; ++o) acc[o] += w[o * 9 + ky * 3 + kx] * xv;
    }
  }
  __bf16* op = out + (size_t)idx * 16;
#pragma unroll
  for (int o = 0; o < 16; ++o) op[o] = (__bf16)acc[o];
}

// ---------------------------------------------------------------------------
// Branchless boundary B load: clamp address (always in-range), AND-mask to 0.
// ---------------------------------------------------------------------------
__device__ __forceinline__ bf16x16 load_b_edge(const char* __restrict__ imgB,
                                               int yi, int xi, int H, int W) {
  const bool valid = (yi >= 0) & (yi < H) & (xi >= 0) & (xi < W);
  int yc = yi < 0 ? 0 : (yi >= H ? H - 1 : yi);
  int xc = xi < 0 ? 0 : (xi >= W ? W - 1 : xi);
  u32x8 bu = *(const u32x8*)(imgB + (yc * W + xc) * 32);
  const unsigned m = valid ? 0xFFFFFFFFu : 0u;
#pragma unroll
  for (int i = 0; i < 8; ++i) bu[i] &= m;
  return __builtin_bit_cast(bf16x16, bu);
}

// ---------------------------------------------------------------------------
// WMMA conv 16->16, 3x3, pad 1, stride 1. bf16 NHWC16 in -> bf16 NHWC out
// (outC = 16, or 64 with channel offset into the concat buffer).
// One wave computes FOUR 16ch x 16x tiles of one output row (A hoisted),
// 20 bf16 WMMAs per wave. Per sub-tile: preload 5 B operands (one load
// clause, staggered waits), then a 5-WMMA chain.
// `hasRes` is a scalar flag so the residual branch stays wave-uniform/SALU.
// ---------------------------------------------------------------------------
__global__ __launch_bounds__(256) void conv16_wmma(
    const __bf16* __restrict__ in,    // NHWC16, H x W
    const __bf16* __restrict__ wpk,   // [10][16][16] bf16
    const float*  __restrict__ bias,  // [16]
    const __bf16* __restrict__ res,   // NHWC16 residual (valid ptr; used iff hasRes)
    __bf16* __restrict__ out,
    int N, int H, int W, int outC, int chOff, int doRelu, int hasRes) {
  const int gpr   = W >> 6;                       // 4-tile groups per row
  const int tiles = N * H * gpr;
  const int wid   = blockIdx.x * (blockDim.x >> 5) + (threadIdx.x >> 5);
  if (wid >= tiles) return;                       // wave-uniform exit
  const int lane = threadIdx.x & 31;
  const int nc   = lane & 15;                     // B/D column; A row m
  const int half = lane >> 4;

  const int gx = (wid % gpr) << 6;
  const int tq = wid / gpr;
  const int y  = tq % H;
  const int n  = tq / H;

  const char* imgB = (const char*)(in  + (size_t)n * H * W * 16);
  const char* resB = (const char*)(res + (size_t)n * H * W * 16);
  char*       outB = (char*)(out + (size_t)n * H * W * outC);

  // Bias slice for this lane half
  float b8[8];
#pragma unroll
  for (int r = 0; r < 8; ++r) b8[r] = bias[half * 8 + r];

  // All 5 A operands, shared across the 4 x-tiles.
  bf16x16 A[5];
#pragma unroll
  for (int p = 0; p < 5; ++p) {
    A[p] = *(const bf16x16*)(wpk + (2 * p + half) * 256 + nc * 16);
  }

  // Per-lane tap geometry (byte offsets); tap 9 aliases tap 8 (A weights 0).
  int tky[5], tkx[5], toff[5];
#pragma unroll
  for (int p = 0; p < 5; ++p) {
    int t = 2 * p + half;
    if (t > 8) t = 8;
    tky[p] = t / 3 - 1;
    tkx[p] = t % 3 - 1;
    toff[p] = (tky[p] * W + tkx[p]) * 32;         // bytes (16 ch x 2B)
  }

  const bool yIn = (y >= 1) && (y < H - 1);

#pragma unroll
  for (int s = 0; s < 4; ++s) {
    const int x0    = gx + s * 16;
    const int xl    = x0 + nc;
    const int pixcB = (y * W + xl) * 32;          // center pixel (bytes)

    // ---- load phase: 5 B operands into distinct registers ----
    bf16x16 Bv[5];
    if (yIn && x0 >= 16 && x0 <= W - 32) {        // wave-uniform interior test
#pragma unroll
      for (int p = 0; p < 5; ++p)
        Bv[p] = *(const bf16x16*)(imgB + (pixcB + toff[p]));
    } else {
#pragma unroll
      for (int p = 0; p < 5; ++p)
        Bv[p] = load_b_edge(imgB, y + tky[p], xl + tkx[p], H, W);
    }

    // ---- 5-WMMA chain ----
    f32x8 acc;
#pragma unroll
    for (int r = 0; r < 8; ++r) acc[r] = b8[r];
#pragma unroll
    for (int p = 0; p < 5; ++p) acc = wmma_bf16(A[p], Bv[p], acc);

    // ---- epilogue ----
    if (doRelu) {
#pragma unroll
      for (int r = 0; r < 8; ++r) acc[r] = acc[r] > 0.0f ? acc[r] : 0.0f;
    }
    if (hasRes) {
      bf16x8 rv = *(const bf16x8*)(resB + pixcB + half * 16);
#pragma unroll
      for (int r = 0; r < 8; ++r) acc[r] += (float)rv[r];
    }
    bf16x8 ov;
#pragma unroll
    for (int r = 0; r < 8; ++r) ov[r] = (__bf16)acc[r];
    *(bf16x8*)(outB + ((y * W + xl) * outC + chOff + half * 8) * 2) = ov;
  }
}

// ---------------------------------------------------------------------------
// WMMA up-block: conv 16->64 + ReLU + pixel_shuffle(2). NHWC16 (H,W) in ->
// NHWC16 (2H,2W) out. 5 B operands hoisted across 4 M-tiles: 20 WMMAs/wave.
// ---------------------------------------------------------------------------
__global__ __launch_bounds__(256) void upconv_wmma(
    const __bf16* __restrict__ in,    // NHWC16, H x W
    const __bf16* __restrict__ wpk,   // [10][64][16] bf16
    const float*  __restrict__ bias,  // [64]
    __bf16* __restrict__ out,         // NHWC16, 2H x 2W
    int N, int H, int W) {
  const int wpr   = W >> 4;
  const int tiles = N * H * wpr;
  const int wid   = blockIdx.x * (blockDim.x >> 5) + (threadIdx.x >> 5);
  if (wid >= tiles) return;
  const int lane = threadIdx.x & 31;
  const int nc   = lane & 15;
  const int half = lane >> 4;

  const int x0 = (wid % wpr) << 4;
  const int tq = wid / wpr;
  const int y  = tq % H;
  const int n  = tq / H;

  const char* imgB = (const char*)(in + (size_t)n * H * W * 16);
  const int xl    = x0 + nc;
  const int pixcB = (y * W + xl) * 32;

  bf16x16 Bv[5];
  if (y >= 1 && y < H - 1 && x0 >= 16 && x0 <= W - 32) {
#pragma unroll
    for (int p = 0; p < 5; ++p) {
      int t = 2 * p + half;
      if (t > 8) t = 8;                            // tap-9 alias (A zeros)
      const int off = ((t / 3 - 1) * W + (t % 3 - 1)) * 32;
      Bv[p] = *(const bf16x16*)(imgB + (pixcB + off));
    }
  } else {
#pragma unroll
    for (int p = 0; p < 5; ++p) {
      int t = 2 * p + half;
      if (t > 8) t = 8;
      Bv[p] = load_b_edge(imgB, y + t / 3 - 1, xl + t % 3 - 1, H, W);
    }
  }

  const int Ho = 2 * H, Wo = 2 * W;
  __bf16* outImg = out + (size_t)n * Ho * Wo * 16;

#pragma unroll
  for (int mt = 0; mt < 4; ++mt) {
    f32x8 acc;
#pragma unroll
    for (int r = 0; r < 8; ++r) acc[r] = bias[mt * 16 + half * 8 + r];
#pragma unroll
    for (int p = 0; p < 5; ++p) {
      bf16x16 A = *(const bf16x16*)(wpk + (2 * p + half) * 1024 + (mt * 16 + nc) * 16);
      acc = wmma_bf16(A, Bv[p], acc);
    }
    // ReLU + pixel shuffle: conv channel q -> out ch q>>2 at (2y+dr, 2x+dc)
#pragma unroll
    for (int r = 0; r < 8; ++r) {
      float v = acc[r] > 0.0f ? acc[r] : 0.0f;
      const int q  = mt * 16 + half * 8 + r;
      const int o  = q >> 2;
      const int dr = (q >> 1) & 1;
      const int dc = q & 1;
      outImg[((2 * y + dr) * Wo + 2 * xl + dc) * 16 + o] = (__bf16)v;
    }
  }
}

// ---------------------------------------------------------------------------
// Tail conv: 64->1 ch, 3x3, pad 1. bf16 NHWC64 in -> fp32 NCHW(1) out.
// Weights staged in LDS.
// ---------------------------------------------------------------------------
__global__ __launch_bounds__(256) void tail_conv(const __bf16* __restrict__ feat,
                                                 const float* __restrict__ tw,  // [64][9]
                                                 const float* __restrict__ tb,  // [1]
                                                 float* __restrict__ out,
                                                 int N, int H, int W) {
  __shared__ float wsm[576];
  for (int i = threadIdx.x; i < 576; i += blockDim.x) wsm[i] = tw[i];
  __syncthreads();

  int idx = blockIdx.x * blockDim.x + threadIdx.x;
  int tot = N * H * W;
  if (idx >= tot) return;
  int xo = idx % W;
  int t  = idx / W;
  int yo = t % H;
  int n  = t / H;

  const __bf16* img = feat + (size_t)n * H * W * 64;
  float acc = tb[0];
  for (int ky = 0; ky < 3; ++ky) {
    int yi = yo + ky - 1;
    if (yi < 0 || yi >= H) continue;
    for (int kx = 0; kx < 3; ++kx) {
      int xi = xo + kx - 1;
      if (xi < 0 || xi >= W) continue;
      const __bf16* base = img + (yi * W + xi) * 64;
      const int tap = ky * 3 + kx;
#pragma unroll
      for (int c8 = 0; c8 < 8; ++c8) {
        bf16x8 v = *(const bf16x8*)(base + c8 * 8);
#pragma unroll
        for (int r = 0; r < 8; ++r)
          acc += (float)v[r] * wsm[(c8 * 8 + r) * 9 + tap];
      }
    }
  }
  out[idx] = acc;
}

// ---------------------------------------------------------------------------
// Host-side orchestration
// ---------------------------------------------------------------------------
static inline void launch_conv16(const __bf16* in, const __bf16* wpk, const float* bias,
                                 const __bf16* res, __bf16* out, int N, int H, int W,
                                 int outC, int chOff, int relu, hipStream_t stream) {
  long tiles = (long)N * H * (W / 64);   // 4-tile groups
  unsigned grid = (unsigned)((tiles + 7) / 8);
  const int hasRes = res != nullptr;
  conv16_wmma<<<grid, 256, 0, stream>>>(in, wpk, bias, hasRes ? res : in, out,
                                        N, H, W, outC, chOff, relu, hasRes);
}

static inline void launch_upconv(const __bf16* in, const __bf16* wpk, const float* bias,
                                 __bf16* out, int N, int H, int W, hipStream_t stream) {
  long tiles = (long)N * H * (W / 16);
  unsigned grid = (unsigned)((tiles + 7) / 8);
  upconv_wmma<<<grid, 256, 0, stream>>>(in, wpk, bias, out, N, H, W);
}

extern "C" void kernel_launch(void* const* d_in, const int* in_sizes, int n_in,
                              void* d_out, int out_size, void* d_ws, size_t ws_size,
                              hipStream_t stream) {
  (void)in_sizes; (void)n_in; (void)out_size; (void)ws_size;
  const float* x      = (const float*)d_in[0];
  const float* head_w = (const float*)d_in[1];   // (4,16,1,3,3)
  const float* head_b = (const float*)d_in[2];   // (4,16)
  const float* res_w  = (const float*)d_in[3];   // (4,4,2,16,16,3,3)
  const float* res_b  = (const float*)d_in[4];   // (4,4,2,16)
  const float* up_w   = (const float*)d_in[5];   // (6,64,16,3,3)
  const float* up_b   = (const float*)d_in[6];   // (6,64)
  const float* out_w  = (const float*)d_in[7];   // (4,16,16,3,3)
  const float* out_b  = (const float*)d_in[8];   // (4,16)
  const float* tail_w = (const float*)d_in[9];   // (1,64,3,3)
  const float* tail_b = (const float*)d_in[10];  // (1,)
  float* y_out = (float*)d_out;                  // (8,1,512,512) fp32

  // Workspace carve-up (~449 MiB + packed weights)
  const size_t CONCAT_BYTES = (size_t)8 * 512 * 512 * 64 * 2;  // 256 MiB bf16 NHWC64
  const size_t BUF_BYTES    = (size_t)8 * 512 * 512 * 16 * 2;  //  64 MiB bf16 NHWC16
  char* ws = (char*)d_ws;
  __bf16* concat = (__bf16*)ws;
  __bf16* bufA   = (__bf16*)(ws + CONCAT_BYTES);
  __bf16* bufB   = (__bf16*)(ws + CONCAT_BYTES + BUF_BYTES);
  __bf16* bufC   = (__bf16*)(ws + CONCAT_BYTES + 2 * BUF_BYTES);
  __bf16* resPk  = (__bf16*)(ws + CONCAT_BYTES + 3 * BUF_BYTES);
  __bf16* outPk  = resPk + 32 * 2560;   // 32 res convs x [10][16][16]
  __bf16* upPk   = outPk + 4 * 2560;    // 4 out convs; then 6 x [10][64][16]

  // Pack all WMMA weights to bf16 [tap10][O][I]
  for (int c = 0; c < 32; ++c)
    pack_w<<<(2560 + 255) / 256, 256, 0, stream>>>(res_w + (long)c * 2304, resPk + (long)c * 2560, 16, 16);
  for (int p = 0; p < 4; ++p)
    pack_w<<<(2560 + 255) / 256, 256, 0, stream>>>(out_w + (long)p * 2304, outPk + (long)p * 2560, 16, 16);
  for (int u = 0; u < 6; ++u)
    pack_w<<<(10240 + 255) / 256, 256, 0, stream>>>(up_w + (long)u * 9216, upPk + (long)u * 10240, 64, 16);

  const int strides[4] = {1, 2, 4, 8};
  const int upFirst[4] = {0, 0, 1, 3};
  const int upCount[4] = {0, 1, 2, 3};

  for (int p = 0; p < 4; ++p) {
    const int s = strides[p];
    int H = 512 / s;

    // Head conv -> bufA (NHWC16, HxH)
    {
      long tot = (long)8 * H * H;
      head_conv<<<(unsigned)((tot + 255) / 256), 256, 0, stream>>>(
          x, head_w + (long)p * 144, head_b + (long)p * 16, bufA, 8, 512, 512, s, H, H);
    }

    // 4 ResBlocks: t = relu(conv(y)); y = y + relu(conv(t))
    __bf16* y = bufA; __bf16* tmp = bufB; __bf16* other = bufC;
    for (int blk = 0; blk < 4; ++blk) {
      const long cidx = ((long)p * 4 + blk) * 2;
      const __bf16* w0 = resPk + (cidx + 0) * 2560;
      const __bf16* w1 = resPk + (cidx + 1) * 2560;
      const float* b0 = res_b + (cidx + 0) * 16;
      const float* b1 = res_b + (cidx + 1) * 16;
      launch_conv16(y, w0, b0, nullptr, tmp, 8, H, H, 16, 0, 1, stream);
      launch_conv16(tmp, w1, b1, y, other, 8, H, H, 16, 0, 1, stream);
      __bf16* sw = y; y = other; other = sw;
    }

    // Up-sampling blocks: conv 16->64 + relu + pixel_shuffle(2)
    for (int k = 0; k < upCount[p]; ++k) {
      const int u = upFirst[p] + k;
      __bf16* dst = (y == bufA) ? bufB : bufA;
      launch_upconv(y, upPk + (long)u * 10240, up_b + (long)u * 64, dst, 8, H, H, stream);
      y = dst;
      H *= 2;
    }

    // Out conv 16->16, written into concat buffer at channel offset 16p
    launch_conv16(y, outPk + (long)p * 2560, out_b + (long)p * 16, nullptr,
                  concat, 8, 512, 512, 64, 16 * p, 0, stream);
  }

  // Tail conv 64->1 -> fp32 output
  {
    long tot = (long)8 * 512 * 512;
    tail_conv<<<(unsigned)((tot + 255) / 256), 256, 0, stream>>>(
        concat, tail_w, tail_b, y_out, 8, 512, 512);
  }
}
